// WanSelfAttention_28956669510080
// MI455X (gfx1250) — matrix operand order repaired and checked by reading, not compiled, not run on hardware
//
#include <hip/hip_runtime.h>
#include <hip/hip_bf16.h>
#include <math.h>

typedef __attribute__((ext_vector_type(16))) _Float16 v16h;
typedef __attribute__((ext_vector_type(8)))  float    v8f;

#define S_TOK 2688
#define DIMSZ 1536
#define NH    12
#define HD    128
#define CC    64          // HD/2 complex pairs
#define EPSV  1e-6f

// Async DMA: global (16B per lane) -> LDS, tracked by ASYNCcnt. ISA §15.18.3 op 98.
// Generic pointers to LDS carry the LDS byte address in bits [31:0] (ISA §10.2).
__device__ __forceinline__ void async_g2l_b128(const void* gsrc, void* ldst)
{
    unsigned loff = (unsigned)(size_t)ldst;
    asm volatile("global_load_async_to_lds_b128 %0, %1, off"
                 :: "v"(loff), "v"(gsrc) : "memory");
}
template <int N>
__device__ __forceinline__ void wait_async_le()
{
    asm volatile("s_wait_asynccnt %0" :: "i"(N) : "memory");
}

// ---------------------------------------------------------------------------
// f32 -> f16 convert (one-time passes over x, weights, and V)
// ---------------------------------------------------------------------------
__global__ void conv_f16(const float* __restrict__ in, _Float16* __restrict__ out, int n)
{
    int i = blockIdx.x * 256 + threadIdx.x;
    if (i < n) out[i] = (_Float16)in[i];
}

// ---------------------------------------------------------------------------
// GEMM: C[M,N] = A[M,K] * B[N,K]^T + bias[N], A/B f16, C f32.
// M=2688, N=K=1536. Block: 256 thr (8 waves), tile 128x128, BK=32.
// Wave w: M-tiles {2*(w&3), 2*(w&3)+1}, N-tiles {4*(w>>2)..+3}
// -> 8 V_WMMA_F32_16X16X32_F16 per K-step per wave.
// Double-buffered GLOBAL_LOAD_ASYNC_TO_LDS_B128: DMA of chunk i+1 overlaps
// WMMA on chunk i; per-wave ASYNCcnt (4 instr/chunk) gates buffer readiness.
// ---------------------------------------------------------------------------
__global__ __launch_bounds__(256)
void gemm_wmma_f16(const _Float16* __restrict__ A, const _Float16* __restrict__ Bm,
                   const float* __restrict__ bias, float* __restrict__ C)
{
    constexpr int K = DIMSZ, N = DIMSZ;
    constexpr int LDA = 40;                       // f16 elems, padded (80B rows)
    constexpr int NI  = K / 32;                   // 48 K-steps
    __shared__ _Float16 As[2][128 * LDA];
    __shared__ _Float16 Bs[2][128 * LDA];

    const int t     = threadIdx.x;
    const int wave  = t >> 5;
    const int lane  = t & 31;
    const int lrow  = lane & 15;
    const int khalf = lane >> 4;

    const int bm  = blockIdx.x * 128;
    const int bn  = blockIdx.y * 128;
    const int tm0 = (wave & 3) * 2;               // 2 M-tiles
    const int tn0 = (wave >> 2) * 4;              // 4 N-tiles

    // per-thread staging coords: 512 b128 groups per matrix, 2 per thread
    const int r0  = t >> 2;                       // row for group j=0 (0..63)
    const int c80 = t & 3;
    auto stage = [&](int it) {
        const int k0  = it * 32;
        const int buf = it & 1;
        #pragma unroll
        for (int j = 0; j < 2; ++j) {
            int r = r0 + j * 64;
            async_g2l_b128(&A [(size_t)(bm + r) * K + k0 + c80 * 8],
                           &As[buf][r * LDA + c80 * 8]);
            async_g2l_b128(&Bm[(size_t)(bn + r) * K + k0 + c80 * 8],
                           &Bs[buf][r * LDA + c80 * 8]);
        }
    };

    v8f acc[2][4];
    #pragma unroll
    for (int i = 0; i < 2; ++i)
        #pragma unroll
        for (int j = 0; j < 4; ++j) acc[i][j] = (v8f){};

    stage(0);
    for (int it = 0; it < NI; ++it) {
        const int buf = it & 1;
        if (it + 1 < NI) { stage(it + 1); wait_async_le<4>(); }
        else             { wait_async_le<0>(); }
        __syncthreads();

        // A fragments (16x32): lane<16 holds K {0..15}, lane>=16 {16..31}
        v16h a[2];
        #pragma unroll
        for (int i = 0; i < 2; ++i)
            #pragma unroll
            for (int e = 0; e < 16; ++e) {
                int kk2 = khalf * 16 + e;
                a[i][e] = As[buf][((tm0 + i) * 16 + lrow) * LDA + kk2];
            }
        // B fragments (32x16): lane col = lane&15, K = khalf*16 + e
        v16h b[4];
        #pragma unroll
        for (int j = 0; j < 4; ++j)
            #pragma unroll
            for (int e = 0; e < 16; ++e)
                b[j][e] = Bs[buf][((tn0 + j) * 16 + lrow) * LDA + khalf * 16 + e];

        #pragma unroll
        for (int i = 0; i < 2; ++i)
            #pragma unroll
            for (int j = 0; j < 4; ++j)
                acc[i][j] = __builtin_amdgcn_wmma_f32_16x16x32_f16(
                    false, a[i], false, b[j], (short)0, acc[i][j], false, false);
        __syncthreads();   // reads done before buf is refilled at it+2
    }

    #pragma unroll
    for (int j = 0; j < 4; ++j) {
        const int n    = bn + (tn0 + j) * 16 + lrow;
        const float bv = bias[n];
        #pragma unroll
        for (int i = 0; i < 2; ++i) {
            const size_t mrow = (size_t)(bm + (tm0 + i) * 16 + khalf * 8) * N + n;
            #pragma unroll
            for (int r2 = 0; r2 < 8; ++r2)
                C[mrow + (size_t)r2 * N] = acc[i][j][r2] + bv;
        }
    }
}

// ---------------------------------------------------------------------------
// RMSNorm (over DIM=1536) + RoPE (per-head C=64 pairs), f32 in -> f16 out.
// One block per token, 256 threads, 3 complex pairs per thread.
// ---------------------------------------------------------------------------
__global__ __launch_bounds__(256)
void rmsnorm_rope(const float* __restrict__ xin, const float* __restrict__ g,
                  const int* __restrict__ grid_sizes,
                  const float* __restrict__ fcos, const float* __restrict__ fsin,
                  _Float16* __restrict__ out)
{
    const int s = blockIdx.x;
    const int t = threadIdx.x;
    const float* row = xin + (size_t)s * DIMSZ;

    float xv[6];
    float ss = 0.f;
    #pragma unroll
    for (int j = 0; j < 3; ++j) {
        int p = t + 256 * j;                    // pair index 0..767
        xv[2*j]   = row[2*p];
        xv[2*j+1] = row[2*p+1];
        ss += xv[2*j]*xv[2*j] + xv[2*j+1]*xv[2*j+1];
    }
    #pragma unroll
    for (int off = 16; off; off >>= 1) ss += __shfl_xor(ss, off, 32);

    __shared__ float wsum[8];
    __shared__ float stot;
    if ((t & 31) == 0) wsum[t >> 5] = ss;
    __syncthreads();
    if (t == 0) {
        float a = 0.f;
        #pragma unroll
        for (int i = 0; i < 8; ++i) a += wsum[i];
        stot = a;
    }
    __syncthreads();
    const float scale = rsqrtf(stot / (float)DIMSZ + EPSV);

    const int hdim = grid_sizes[1], wdim = grid_sizes[2];
    const int zf = s / (hdim * wdim);
    const int yh = (s / wdim) % hdim;
    const int xw = s % wdim;
    const int s1 = CC - 2 * (CC / 3);           // 22
    const int s2 = CC / 3;                      // 21

    _Float16* orow = out + (size_t)s * DIMSZ;
    #pragma unroll
    for (int j = 0; j < 3; ++j) {
        int p = t + 256 * j;
        int c = p & (CC - 1);                   // pair within head
        int pos = (c < s1) ? zf : ((c < s1 + s2) ? yh : xw);
        float cv = fcos[pos * CC + c];
        float sv = fsin[pos * CC + c];
        float y0 = xv[2*j]   * scale * g[2*p];
        float y1 = xv[2*j+1] * scale * g[2*p+1];
        orow[2*p]   = (_Float16)(y0 * cv - y1 * sv);
        orow[2*p+1] = (_Float16)(y0 * sv + y1 * cv);
    }
}

// ---------------------------------------------------------------------------
// Flash attention: scale*softmax(QK^T)V per head, D=128, f16 in/out.
// Block: 128 thr (4 waves), head = blockIdx.y, 64 queries per block
// (wave w owns 16). K/V chunks (32 keys) double-buffered via async DMA
// (8 instr/chunk/wave); online softmax; P transposed through per-wave LDS.
// ---------------------------------------------------------------------------
__global__ __launch_bounds__(128)
void flash_attn(const _Float16* __restrict__ q, const _Float16* __restrict__ k,
                const _Float16* __restrict__ v, const int* __restrict__ seq_lens,
                _Float16* __restrict__ o)
{
    constexpr int NC = S_TOK / 32;              // 84 key chunks
    __shared__ _Float16 Ks[2][32 * HD];
    __shared__ _Float16 Vs[2][32 * HD];
    __shared__ _Float16 Pb[4][16 * 32];

    const int t     = threadIdx.x;
    const int wave  = t >> 5;
    const int lane  = t & 31;
    const int lrow  = lane & 15;
    const int khalf = lane >> 4;

    const int head   = blockIdx.y;
    const int q0     = blockIdx.x * 64 + wave * 16;
    const int seqlen = seq_lens[0];
    const size_t hoff = (size_t)head * HD;
    const float scale = 0.08838834764831845f;   // 1/sqrt(128)

    // staging coords: 512 b128 groups per matrix, 4 per thread per matrix
    const int rr  = t >> 2;                     // key row 0..31
    const int cc0 = t & 3;                      // base 8-f16 group
    auto stage = [&](int it) {
        const int kb  = it * 32;
        const int buf = it & 1;
        #pragma unroll
        for (int h2 = 0; h2 < 4; ++h2) {
            int c8 = cc0 * 4 + h2;              // 0..15
            async_g2l_b128(&k[(size_t)(kb + rr) * DIMSZ + hoff + c8 * 8],
                           &Ks[buf][rr * HD + c8 * 8]);
            async_g2l_b128(&v[(size_t)(kb + rr) * DIMSZ + hoff + c8 * 8],
                           &Vs[buf][rr * HD + c8 * 8]);
        }
    };

    // Q A-fragments for 4 chunks of d (resident whole kernel)
    v16h qa[4];
    #pragma unroll
    for (int kc = 0; kc < 4; ++kc)
        #pragma unroll
        for (int e = 0; e < 16; ++e) {
            int d = kc * 32 + khalf * 16 + e;
            qa[kc][e] = q[(size_t)(q0 + lrow) * DIMSZ + hoff + d];
        }

    v8f zero = {};
    v8f oacc[8];
    #pragma unroll
    for (int i = 0; i < 8; ++i) oacc[i] = zero;
    float m_r[8], l_r[8];
    #pragma unroll
    for (int i = 0; i < 8; ++i) { m_r[i] = -__builtin_inff(); l_r[i] = 0.f; }

    stage(0);
    for (int it = 0; it < NC; ++it) {
        const int kb  = it * 32;
        const int buf = it & 1;
        if (it + 1 < NC) { stage(it + 1); wait_async_le<8>(); }
        else             { wait_async_le<0>(); }
        __syncthreads();

        // scores: two 16x16 key tiles
        v8f s0 = zero, s1 = zero;
        #pragma unroll
        for (int kc = 0; kc < 4; ++kc) {
            v16h b0, b1;
            #pragma unroll
            for (int e = 0; e < 16; ++e) {
                int d = kc * 32 + khalf * 16 + e;
                b0[e] = Ks[buf][(lrow     ) * HD + d];
                b1[e] = Ks[buf][(16 + lrow) * HD + d];
            }
            s0 = __builtin_amdgcn_wmma_f32_16x16x32_f16(false, qa[kc], false, b0,
                                                        (short)0, s0, false, false);
            s1 = __builtin_amdgcn_wmma_f32_16x16x32_f16(false, qa[kc], false, b1,
                                                        (short)0, s1, false, false);
        }
        const bool valid0 = (kb + lrow)      < seqlen;
        const bool valid1 = (kb + 16 + lrow) < seqlen;
        #pragma unroll
        for (int r = 0; r < 8; ++r) {
            s0[r] = valid0 ? s0[r] * scale : -__builtin_inff();
            s1[r] = valid1 ? s1[r] * scale : -__builtin_inff();
        }

        // online softmax per query row (rows in VGPR index, key cols in lanes)
        float p0x[8], p1x[8];
        #pragma unroll
        for (int r = 0; r < 8; ++r) {
            float mx = fmaxf(s0[r], s1[r]);
            #pragma unroll
            for (int off = 8; off; off >>= 1) mx = fmaxf(mx, __shfl_xor(mx, off, 32));
            float mnew = fmaxf(m_r[r], mx);
            float p0 = __expf(s0[r] - mnew);
            float p1 = __expf(s1[r] - mnew);
            float sum = p0 + p1;
            #pragma unroll
            for (int off = 8; off; off >>= 1) sum += __shfl_xor(sum, off, 32);
            float corr = __expf(m_r[r] - mnew);
            l_r[r] = l_r[r] * corr + sum;
            m_r[r] = mnew;
            p0x[r] = p0; p1x[r] = p1;
            #pragma unroll
            for (int nt = 0; nt < 8; ++nt) oacc[nt][r] *= corr;
        }

        // transpose P (D-layout) -> A-layout via per-wave LDS buffer
        #pragma unroll
        for (int r = 0; r < 8; ++r) {
            Pb[wave][(khalf * 8 + r) * 32 + lrow]      = (_Float16)p0x[r];
            Pb[wave][(khalf * 8 + r) * 32 + 16 + lrow] = (_Float16)p1x[r];
        }
        asm volatile("s_wait_dscnt 0x0" ::: "memory");
        v16h pa;
        #pragma unroll
        for (int e = 0; e < 16; ++e) {
            int kk = khalf * 16 + e;
            pa[e] = Pb[wave][lrow * 32 + kk];
        }

        // P (16x32) @ V (32x128): 8 output d-tiles
        #pragma unroll
        for (int nt = 0; nt < 8; ++nt) {
            v16h bv;
            #pragma unroll
            for (int e = 0; e < 16; ++e)
                bv[e] = Vs[buf][(khalf * 16 + e) * HD + nt * 16 + lrow];
            oacc[nt] = __builtin_amdgcn_wmma_f32_16x16x32_f16(false, pa, false, bv,
                                                              (short)0, oacc[nt], false, false);
        }
        __syncthreads();   // reads done before buf is refilled at it+2
    }

    // epilogue: divide by denominator, write f16 (S, N, D)
    #pragma unroll
    for (int nt = 0; nt < 8; ++nt)
        #pragma unroll
        for (int r = 0; r < 8; ++r) {
            int qrow = q0 + khalf * 8 + r;
            int d    = nt * 16 + lrow;
            o[(size_t)qrow * DIMSZ + hoff + d] = (_Float16)(oacc[nt][r] / l_r[r]);
        }
}

// ---------------------------------------------------------------------------
extern "C" void kernel_launch(void* const* d_in, const int* in_sizes, int n_in,
                              void* d_out, int out_size, void* d_ws, size_t ws_size,
                              hipStream_t stream)
{
    (void)in_sizes; (void)n_in; (void)out_size; (void)ws_size;
    const float* x    = (const float*)d_in[0];
    const int*   seq  = (const int*)d_in[1];
    const int*   gs   = (const int*)d_in[2];
    const float* fcos = (const float*)d_in[3];
    const float* fsin = (const float*)d_in[4];
    const float* Wq   = (const float*)d_in[5];
    const float* bq   = (const float*)d_in[6];
    const float* Wk   = (const float*)d_in[7];
    const float* bk   = (const float*)d_in[8];
    const float* Wv   = (const float*)d_in[9];
    const float* bv   = (const float*)d_in[10];
    const float* Wo   = (const float*)d_in[11];
    const float* bo   = (const float*)d_in[12];
    const float* gq   = (const float*)d_in[13];
    const float* gk   = (const float*)d_in[14];

    char* ws = (char*)d_ws;
    const size_t SM4 = (size_t)S_TOK * DIMSZ * 4;   // f32 activation plane
    const size_t SM2 = (size_t)S_TOK * DIMSZ * 2;   // f16 activation plane
    const size_t WM2 = (size_t)DIMSZ * DIMSZ * 2;   // f16 weight plane
    size_t off = 0;
    float*    qlin = (float*)(ws + off);            off += SM4;
    float*    klin = (float*)(ws + off);            off += SM4;
    float*    vlin = (float*)(ws + off);            off += SM4;
    _Float16* x16  = (_Float16*)(ws + off);         off += SM2;
    _Float16* q16  = (_Float16*)(ws + off);         off += SM2;
    _Float16* k16  = (_Float16*)(ws + off);         off += SM2;
    _Float16* v16  = (_Float16*)(ws + off);         off += SM2;
    _Float16* wq16 = (_Float16*)(ws + off);         off += WM2;
    _Float16* wk16 = (_Float16*)(ws + off);         off += WM2;
    _Float16* wv16 = (_Float16*)(ws + off);         off += WM2;
    _Float16* wo16 = (_Float16*)(ws + off);         off += WM2;
    _Float16* attn16 = x16;                         // x16 dead after QKV GEMMs

    const int nact = S_TOK * DIMSZ;
    const int nwgt = DIMSZ * DIMSZ;
    conv_f16<<<(nact + 255) / 256, 256, 0, stream>>>(x,  x16,  nact);
    conv_f16<<<(nwgt + 255) / 256, 256, 0, stream>>>(Wq, wq16, nwgt);
    conv_f16<<<(nwgt + 255) / 256, 256, 0, stream>>>(Wk, wk16, nwgt);
    conv_f16<<<(nwgt + 255) / 256, 256, 0, stream>>>(Wv, wv16, nwgt);
    conv_f16<<<(nwgt + 255) / 256, 256, 0, stream>>>(Wo, wo16, nwgt);

    dim3 gg(S_TOK / 128, DIMSZ / 128);               // 21 x 12
    gemm_wmma_f16<<<gg, 256, 0, stream>>>(x16, wq16, bq, qlin);
    gemm_wmma_f16<<<gg, 256, 0, stream>>>(x16, wk16, bk, klin);
    gemm_wmma_f16<<<gg, 256, 0, stream>>>(x16, wv16, bv, vlin);

    rmsnorm_rope<<<S_TOK, 256, 0, stream>>>(qlin, gq, gs, fcos, fsin, q16);
    rmsnorm_rope<<<S_TOK, 256, 0, stream>>>(klin, gk, gs, fcos, fsin, k16);
    conv_f16<<<(nact + 255) / 256, 256, 0, stream>>>(vlin, v16, nact);

    flash_attn<<<dim3(S_TOK / 64, NH), 128, 0, stream>>>(q16, k16, v16, seq, attn16);

    gemm_wmma_f16<<<gg, 256, 0, stream>>>(attn16, wo16, bo, (float*)d_out);
}
